// SimpleSNN_50319836840756
// MI455X (gfx1250) — matrix operand order repaired and checked by reading, not compiled, not run on hardware
//
#include <hip/hip_runtime.h>
#include <hip/hip_bf16.h>

#define VOCAB   32000
#define D_MODEL 512
#define HIDDEN  1024
#define BSZ     4
#define TLEN    512
#define ROWS    (BSZ * TLEN)      // 2048
#define GCOLS   64                // W2 columns staged per workgroup
#define LDK     (HIDDEN + 8)      // padded K stride in LDS (bank spread)
#define MBLK    4                 // M-tiles per K-sweep (independent acc chains)

typedef __attribute__((ext_vector_type(16))) __bf16 v16bf;
typedef __attribute__((ext_vector_type(8)))  float  v8f;
typedef __attribute__((ext_vector_type(2)))  float  v2f;

static __device__ __forceinline__ unsigned short f32_to_bf16_rne(float f) {
    unsigned u = __float_as_uint(f);
    unsigned r = u + 0x7FFFu + ((u >> 16) & 1u);   // round-to-nearest-even
    return (unsigned short)(r >> 16);
}
static __device__ __forceinline__ float bf16_bits_to_f32(unsigned short h) {
    return __uint_as_float(((unsigned)h) << 16);
}

// ---------------------------------------------------------------------------
// Kernel 1: X[row][d] = emb[ids[row]][d] + pos[t][d],  row = b*T + t
// ---------------------------------------------------------------------------
__global__ void snn_embed_kernel(const int* __restrict__ ids,
                                 const float* __restrict__ emb,
                                 const float* __restrict__ pos,
                                 float* __restrict__ X) {
    int row = blockIdx.x;                  // 0..2047
    int t   = row & (TLEN - 1);
    long long tok = ids[row];
    const float* e = emb + tok * D_MODEL;
    const float* p = pos + (long long)t * D_MODEL;
    float* x = X + (long long)row * D_MODEL;
    for (int d = threadIdx.x; d < D_MODEL; d += blockDim.x)
        x[d] = e[d] + p[d];
}

// ---------------------------------------------------------------------------
// Kernel 2: h = X * W1 + b1 using exact-precision V_WMMA_F32_16X16X4_F32.
// One wave per 16x16 output tile; 128 M-tiles x 64 N-tiles = 8192 waves.
// ---------------------------------------------------------------------------
__global__ __launch_bounds__(256) void snn_fc1_wmma_kernel(
        const float* __restrict__ X, const float* __restrict__ W1,
        const float* __restrict__ b1, float* __restrict__ H) {
    int wave = (blockIdx.x * blockDim.x + threadIdx.x) >> 5;
    int lane = threadIdx.x & 31;
    int mt = wave & 127;                   // M tile
    int nt = wave >> 7;                    // N tile (0..63)
    int m0 = mt << 4, n0 = nt << 4;
    int half = lane >> 4;                  // lane group 0/1
    int l16  = lane & 15;
    int mA = m0 + l16;                     // A row for this lane
    int nB = n0 + l16;                     // B column for this lane

    const float* xrow = X + (long long)mA * D_MODEL;
    v8f acc = {};
    for (int k0 = 0; k0 < D_MODEL; k0 += 4) {
        int kk = k0 + (half << 1);         // 32-bit A 16x4: lanes>=16 hold K=2,3
        v2f a; a[0] = xrow[kk]; a[1] = xrow[kk + 1];
        v2f b; b[0] = W1[(long long)kk * HIDDEN + nB];
               b[1] = W1[(long long)(kk + 1) * HIDDEN + nB];
        acc = __builtin_amdgcn_wmma_f32_16x16x4_f32(
                  false, a, false, b, (short)0, acc, false, false);
    }
    float bias = b1[nB];
#pragma unroll
    for (int r = 0; r < 8; ++r) {
        int m = m0 + r + (half << 3);      // C/D layout: VGPR r -> M = r (+8)
        H[(long long)m * HIDDEN + nB] = acc[r] + bias;
    }
}

// ---------------------------------------------------------------------------
// Kernel 3: LIF scan over T. One thread per (b,h): 4096 chains, coalesced in h.
// Spikes stored as bf16 bits (exact 0.0 / 1.0) at [b*T+t][h] for GEMM2's A.
// Also writes the two scalar tail outputs (avg_spikes = mem_out = 0).
// ---------------------------------------------------------------------------
__global__ void snn_scan_kernel(const float* __restrict__ H,
                                const float* __restrict__ tau,
                                const float* __restrict__ bthr,
                                const float* __restrict__ athr,
                                unsigned short* __restrict__ spk,
                                float* __restrict__ out_tail) {
    int tid = blockIdx.x * blockDim.x + threadIdx.x;   // 0..4095
    if (tid < 2) out_tail[tid] = 0.0f;
    int b  = tid >> 10;
    int hh = tid & (HIDDEN - 1);
    float decay = __expf(-1.0f / fmaxf(tau[hh], 1.0f));
    float thr   = bthr[hh] + athr[hh];
    float mem = 0.0f;
    long long base = (long long)b * TLEN * HIDDEN + hh;
    for (int t = 0; t < TLEN; ++t) {
        float m = mem * decay + H[base + (long long)t * HIDDEN];
        bool s = (m - thr) >= 0.0f;
        spk[base + (long long)t * HIDDEN] = s ? (unsigned short)0x3F80u
                                              : (unsigned short)0u;
        mem = s ? 0.0f : m;
    }
}

// ---------------------------------------------------------------------------
// Kernel 4: logits = spikes * W2 + b2 via split-precision bf16 WMMA.
// Workgroup = 4 waves, 64 W2 columns staged as bf16 hi+lo (transposed [c][k])
// in 258 KB dynamic LDS (one WG per WGP). Each wave owns 16 columns and
// sweeps MBLK=4 M-tiles per K-pass: per k-chunk, B (hi+lo) is read from LDS
// once and 8 WMMAs are issued across 4 independent accumulator chains, so a
// single wave per SIMD still fills the matrix pipe.
// ---------------------------------------------------------------------------
__global__ __launch_bounds__(128) void snn_fc2_wmma_kernel(
        const unsigned short* __restrict__ spk, const float* __restrict__ W2,
        const float* __restrict__ b2, float* __restrict__ out) {
    extern __shared__ unsigned short lds[];
    unsigned short* hi = lds;
    unsigned short* lo = lds + GCOLS * LDK;
    int nbase = blockIdx.x * GCOLS;

    // Stage + convert W2 columns [nbase, nbase+64): coalesced f32 reads,
    // split into bf16 hi + bf16 residual, stored column-major in LDS.
    for (int idx = threadIdx.x; idx < GCOLS * HIDDEN; idx += blockDim.x) {
        int c = idx & (GCOLS - 1);
        int k = idx >> 6;
        float w = W2[(long long)k * VOCAB + nbase + c];
        unsigned short whi = f32_to_bf16_rne(w);
        unsigned short wlo = f32_to_bf16_rne(w - bf16_bits_to_f32(whi));
        hi[c * LDK + k] = whi;
        lo[c * LDK + k] = wlo;
    }
    __syncthreads();

    int w    = threadIdx.x >> 5;
    int lane = threadIdx.x & 31;
    int half = lane >> 4;
    int l16  = lane & 15;
    int c = (w << 4) + l16;                // LDS column owned by this lane
    int n = nbase + c;                     // global output column
    float bias = b2[n];

    // B 32x16 bf16 layout: lane holds 16 contiguous K for its column,
    // K-half selected by lane group.
    const unsigned short* bhp = hi + c * LDK + (half << 4);
    const unsigned short* blp = lo + c * LDK + (half << 4);

    for (int mt4 = 0; mt4 < ROWS / (16 * MBLK); ++mt4) {   // 32 M-blocks
        int m0 = mt4 * (16 * MBLK);
        // A 16x32 bf16 layout: lane row = m0+l16, 16 contiguous K at
        // k0 + 16*half.
        const unsigned short* arow0 =
            spk + (long long)(m0 + l16) * HIDDEN + (half << 4);

        v8f acc[MBLK];
#pragma unroll
        for (int i = 0; i < MBLK; ++i) acc[i] = (v8f){};

        for (int kc = 0; kc < HIDDEN / 32; ++kc) {
            int k0 = kc << 5;
            union { v16bf v; int4 q[2]; } bh, bl;
            bh.q[0] = *(const int4*)(bhp + k0);
            bh.q[1] = *(const int4*)(bhp + k0 + 8);
            bl.q[0] = *(const int4*)(blp + k0);
            bl.q[1] = *(const int4*)(blp + k0 + 8);
#pragma unroll
            for (int i = 0; i < MBLK; ++i) {
                const unsigned short* ar = arow0 + (long long)i * 16 * HIDDEN;
                union { v16bf v; int4 q[2]; } a;
                a.q[0] = *(const int4*)(ar + k0);
                a.q[1] = *(const int4*)(ar + k0 + 8);
                acc[i] = __builtin_amdgcn_wmma_f32_16x16x32_bf16(
                             false, a.v, false, bh.v, (short)0, acc[i],
                             false, false);
                acc[i] = __builtin_amdgcn_wmma_f32_16x16x32_bf16(
                             false, a.v, false, bl.v, (short)0, acc[i],
                             false, false);
            }
        }
#pragma unroll
        for (int i = 0; i < MBLK; ++i) {
            float* orow = out + (long long)(m0 + i * 16) * VOCAB + n;
#pragma unroll
            for (int r = 0; r < 8; ++r)
                orow[(long long)(r + (half << 3)) * VOCAB] = acc[i][r] + bias;
        }
    }
}

// ---------------------------------------------------------------------------
extern "C" void kernel_launch(void* const* d_in, const int* in_sizes, int n_in,
                              void* d_out, int out_size, void* d_ws, size_t ws_size,
                              hipStream_t stream) {
    const int*   ids  = (const int*)  d_in[0];
    const float* emb  = (const float*)d_in[1];
    const float* pos  = (const float*)d_in[2];
    const float* W1   = (const float*)d_in[3];
    const float* b1   = (const float*)d_in[4];
    const float* W2   = (const float*)d_in[5];
    const float* b2   = (const float*)d_in[6];
    const float* tau  = (const float*)d_in[7];
    const float* bthr = (const float*)d_in[8];
    const float* athr = (const float*)d_in[9];
    float* out = (float*)d_out;

    char* ws = (char*)d_ws;
    float* X  = (float*)ws;                                            // 4 MB
    float* Hb = (float*)(ws + (size_t)ROWS * D_MODEL * sizeof(float)); // 8 MB
    unsigned short* spk = (unsigned short*)(ws +
        (size_t)ROWS * D_MODEL * sizeof(float) +
        (size_t)ROWS * HIDDEN * sizeof(float));                        // 4 MB

    snn_embed_kernel<<<ROWS, 256, 0, stream>>>(ids, emb, pos, X);
    snn_fc1_wmma_kernel<<<(128 * 64) / 8, 256, 0, stream>>>(X, W1, b1, Hb);
    snn_scan_kernel<<<16, 256, 0, stream>>>(Hb, tau, bthr, athr, spk,
                                            out + (size_t)ROWS * VOCAB);
    size_t ldsBytes = 2ull * GCOLS * LDK * sizeof(unsigned short);     // 258 KB
    snn_fc2_wmma_kernel<<<VOCAB / GCOLS, 128, ldsBytes, stream>>>(spk, W2, b2, out);
}
